// ManifoldEmbLoss_20409684591015
// MI455X (gfx1250) — compile-verified, matrix operand order checked by reading, not run on hardware
//
#include <hip/hip_runtime.h>

typedef __bf16 bf16_t;
typedef __attribute__((ext_vector_type(16))) __bf16 v16bf;
typedef __attribute__((ext_vector_type(8)))  float  v8f;

#define N_PTS 8192
#define DIMX  256
#define DIMZ  64
#define KNN   10
#define BIGF  3.4e38f

// workspace byte offsets
#define XSQ_OFF  0u
#define XHI_OFF  32768u                    // 8192 f32
#define XLO_OFF  (XHI_OFF + 4194304u)      // 8192*256 bf16
#define IDX_OFF  (XLO_OFF + 4194304u)      // 8192*256 bf16
#define PART_OFF (IDX_OFF + 327680u)       // 8192*10 i32

// ---------------------------------------------------------------------------
// K1: per-row sum of squares + bf16 hi/lo split of X (one wave per row)
// ---------------------------------------------------------------------------
__global__ void prep_kernel(const float* __restrict__ X, float* __restrict__ Xsq,
                            bf16_t* __restrict__ Xhi, bf16_t* __restrict__ Xlo) {
  int wave = (blockIdx.x * blockDim.x + threadIdx.x) >> 5;
  int lane = threadIdx.x & 31;
  if (wave >= N_PTS) return;
  const float* row = X + (size_t)wave * DIMX;
  float ss = 0.f;
#pragma unroll
  for (int t = 0; t < DIMX / 32; ++t) {
    int d = lane + 32 * t;
    float x = row[d];
    ss += x * x;
    bf16_t h = (bf16_t)x;
    float lo = x - (float)h;
    Xhi[(size_t)wave * DIMX + d] = h;
    Xlo[(size_t)wave * DIMX + d] = (bf16_t)lo;
  }
#pragma unroll
  for (int m = 16; m >= 1; m >>= 1) ss += __shfl_xor(ss, m);
  if (lane == 0) Xsq[wave] = ss;
}

// ---------------------------------------------------------------------------
// K2: fused bf16x3 WMMA Gram tiles + per-row top-10 (self excluded).
// 1 workgroup = 16 rows; 4 waves split the 512 column tiles; merge via LDS.
// ---------------------------------------------------------------------------
__global__ __launch_bounds__(128) void knn_kernel(const bf16_t* __restrict__ Xhi,
                                                  const bf16_t* __restrict__ Xlo,
                                                  const float* __restrict__ Xsq,
                                                  int* __restrict__ idxOut) {
  __shared__ float sTile[4][16][16];
  __shared__ float mVal[16][4][KNN];
  __shared__ int   mIdx[16][4][KNN];

  const int wv   = threadIdx.x >> 5;
  const int lane = threadIdx.x & 31;
  const int half = lane >> 4;        // which half-wave (K-split in 16-bit frag layout)
  const int l16  = lane & 15;
  const int i0   = blockIdx.x * 16;
  const int irow = i0 + l16;

  // Load A fragments for rows i0..i0+15, all K=0..255, hi & lo.
  // CDNA5 16-bit A layout: lane<16 holds row l16 with K = k0+{0..7,16..23},
  // lane>=16 holds K = k0+{8..15,24..31} -> two contiguous 16B runs per frag.
  v16bf ahi[8], alo[8];
  {
    const size_t abase = (size_t)irow * DIMX;
#pragma unroll
    for (int kc = 0; kc < 8; ++kc) {
      int f = kc * 32 + 8 * half;
      union { v16bf bf; uint4 u[2]; } ua, ul;
      ua.u[0] = *(const uint4*)(Xhi + abase + f);
      ua.u[1] = *(const uint4*)(Xhi + abase + f + 16);
      ul.u[0] = *(const uint4*)(Xlo + abase + f);
      ul.u[1] = *(const uint4*)(Xlo + abase + f + 16);
      ahi[kc] = ua.bf;
      alo[kc] = ul.bf;
    }
  }

  float tv[KNN]; int ti[KNN];
#pragma unroll
  for (int r = 0; r < KNN; ++r) { tv[r] = BIGF; ti[r] = 0; }

  for (int t = wv; t < N_PTS / 16; t += 4) {
    const int j0 = t * 16;
    const size_t bbase = (size_t)(j0 + l16) * DIMX;   // B column n=l16 is X row j0+n
    v8f c = {};
#pragma unroll
    for (int kc = 0; kc < 8; ++kc) {
      int f = kc * 32 + 8 * half;
      union { v16bf bf; uint4 u[2]; } ub, vb;
      ub.u[0] = *(const uint4*)(Xhi + bbase + f);
      ub.u[1] = *(const uint4*)(Xhi + bbase + f + 16);
      vb.u[0] = *(const uint4*)(Xlo + bbase + f);
      vb.u[1] = *(const uint4*)(Xlo + bbase + f + 16);
      // bf16x3 error-corrected accumulation: hi*hi + hi*lo + lo*hi
      c = __builtin_amdgcn_wmma_f32_16x16x32_bf16(false, ahi[kc], false, ub.bf,
                                                  (short)0, c, false, false);
      c = __builtin_amdgcn_wmma_f32_16x16x32_bf16(false, ahi[kc], false, vb.bf,
                                                  (short)0, c, false, false);
      c = __builtin_amdgcn_wmma_f32_16x16x32_bf16(false, alo[kc], false, ub.bf,
                                                  (short)0, c, false, false);
    }
    // s[j] = Xsq[j] - 2*G[i,j]  (row-constant Xsq[i] irrelevant for ranking)
    // C layout: vgpr v, lane L -> (row v + 8*(L>=16), col L%16)
    float xsqj = Xsq[j0 + l16];
#pragma unroll
    for (int v = 0; v < 8; ++v)
      sTile[wv][v + 8 * half][l16] = xsqj - 2.0f * c[v];
    __syncthreads();

    if (half == 0) {          // lanes 0..15 each scan their own row
      float worst = tv[KNN - 1];
#pragma unroll 1
      for (int n = 0; n < 16; ++n) {
        int j = j0 + n;
        float s = sTile[wv][l16][n];
        if (j != irow && s < worst) {
          float cv = s; int ci = j;
#pragma unroll
          for (int q = 0; q < KNN; ++q) {      // sorted-insert bubble (registers)
            bool lt = cv < tv[q];
            float nv = lt ? cv : tv[q]; int ni = lt ? ci : ti[q];
            cv = lt ? tv[q] : cv;       ci = lt ? ti[q] : ci;
            tv[q] = nv; ti[q] = ni;
          }
          worst = tv[KNN - 1];
        }
      }
    }
    __syncthreads();
  }

  // merge per-wave top-10 lists
  if (half == 0) {
#pragma unroll
    for (int r = 0; r < KNN; ++r) { mVal[l16][wv][r] = tv[r]; mIdx[l16][wv][r] = ti[r]; }
  }
  __syncthreads();
  if (wv == 0 && half == 0) {
    float fv[KNN]; int fi[KNN];
#pragma unroll
    for (int r = 0; r < KNN; ++r) { fv[r] = BIGF; fi[r] = 0; }
#pragma unroll 1
    for (int w = 0; w < 4; ++w) {
#pragma unroll
      for (int r = 0; r < KNN; ++r) {
        float cv = mVal[l16][w][r]; int ci = mIdx[l16][w][r];
        if (cv < fv[KNN - 1]) {
#pragma unroll
          for (int q = 0; q < KNN; ++q) {
            bool lt = cv < fv[q];
            float nv = lt ? cv : fv[q]; int ni = lt ? ci : fi[q];
            cv = lt ? fv[q] : cv;       ci = lt ? fi[q] : ci;
            fv[q] = nv; fi[q] = ni;
          }
        }
      }
    }
#pragma unroll
    for (int r = 0; r < KNN; ++r) idxOut[(size_t)irow * KNN + r] = fi[r];
  }
}

// ---------------------------------------------------------------------------
// K3: per-row neighbor distances, row-max normalize, per-row |zd - xd| sum.
// One wave per row i; cross-lane reduce with wave32 shuffles.
// ---------------------------------------------------------------------------
__global__ void dist_kernel(const float* __restrict__ z, const float* __restrict__ X,
                            const int* __restrict__ idx, float* __restrict__ partial) {
  int i    = (blockIdx.x * blockDim.x + threadIdx.x) >> 5;
  int lane = threadIdx.x & 31;
  if (i >= N_PTS) return;
  float xd[KNN], zd[KNN];
#pragma unroll
  for (int r = 0; r < KNN; ++r) {
    int j = idx[(size_t)i * KNN + r];
    float sx = 0.f;
#pragma unroll
    for (int t = 0; t < DIMX / 32; ++t) {
      int d = lane + 32 * t;
      float dif = X[(size_t)i * DIMX + d] - X[(size_t)j * DIMX + d];
      sx += dif * dif;
    }
    float sz = 0.f;
#pragma unroll
    for (int t = 0; t < DIMZ / 32; ++t) {
      int d = lane + 32 * t;
      float dif = z[(size_t)i * DIMZ + d] - z[(size_t)j * DIMZ + d];
      sz += dif * dif;
    }
#pragma unroll
    for (int m = 16; m >= 1; m >>= 1) { sx += __shfl_xor(sx, m); sz += __shfl_xor(sz, m); }
    xd[r] = sqrtf(sx);
    zd[r] = sqrtf(sz);
  }
  float mx = 1e-8f, mz = 1e-8f;
#pragma unroll
  for (int r = 0; r < KNN; ++r) { mx = fmaxf(mx, xd[r]); mz = fmaxf(mz, zd[r]); }
  float s = 0.f;
#pragma unroll
  for (int r = 0; r < KNN; ++r) s += fabsf(zd[r] / mz - xd[r] / mx);
  if (lane == 0) partial[i] = s;
}

// ---------------------------------------------------------------------------
// K4: deterministic final reduction -> mean
// ---------------------------------------------------------------------------
__global__ void reduce_kernel(const float* __restrict__ partial, float* __restrict__ out) {
  __shared__ float red[256];
  float s = 0.f;
  for (int i = threadIdx.x; i < N_PTS; i += 256) s += partial[i];
  red[threadIdx.x] = s;
  __syncthreads();
  for (int m = 128; m >= 1; m >>= 1) {
    if ((int)threadIdx.x < m) red[threadIdx.x] += red[threadIdx.x + m];
    __syncthreads();
  }
  if (threadIdx.x == 0) out[0] = red[0] / (float)(N_PTS * KNN);
}

extern "C" void kernel_launch(void* const* d_in, const int* in_sizes, int n_in,
                              void* d_out, int out_size, void* d_ws, size_t ws_size,
                              hipStream_t stream) {
  const float* z = (const float*)d_in[0];   // (8192, 64)
  const float* X = (const float*)d_in[1];   // (8192, 256)
  char* ws = (char*)d_ws;
  float*  Xsq     = (float*)(ws + XSQ_OFF);
  bf16_t* Xhi     = (bf16_t*)(ws + XHI_OFF);
  bf16_t* Xlo     = (bf16_t*)(ws + XLO_OFF);
  int*    idx     = (int*)(ws + IDX_OFF);
  float*  partial = (float*)(ws + PART_OFF);
  float*  out     = (float*)d_out;

  prep_kernel<<<1024, 256, 0, stream>>>(X, Xsq, Xhi, Xlo);
  knn_kernel<<<N_PTS / 16, 128, 0, stream>>>(Xhi, Xlo, Xsq, idx);
  dist_kernel<<<1024, 256, 0, stream>>>(z, X, idx, partial);
  reduce_kernel<<<1, 256, 0, stream>>>(partial, out);
}
